// Router_58969900974343
// MI455X (gfx1250) — matrix operand order repaired and checked by reading, not compiled, not run on hardware
//
#include <hip/hip_runtime.h>
#include <hip/hip_bf16.h>

typedef __attribute__((ext_vector_type(16))) __bf16 v16bf;
typedef __attribute__((ext_vector_type(8)))  float  v8f;

#define D      2048
#define NEXP   64
#define NTOK   16384
#define APITCH 2056          // 2048 + 8 bf16 (16B) pad -> bank rotation, keeps 16B align
#define LN_EPS 1e-5f

static __device__ __forceinline__ unsigned short f2bf(float f) {
    union { float f; unsigned u; } c; c.f = f;
    unsigned u = c.u;
    u += 0x7fffu + ((u >> 16) & 1u);   // round-to-nearest-even
    return (unsigned short)(u >> 16);
}

// monotone float -> uint mapping (preserves total order incl. negatives)
static __device__ __forceinline__ unsigned ford(float f) {
    union { float f; unsigned u; } c; c.f = f;
    return c.u ^ ((c.u >> 31) ? 0xFFFFFFFFu : 0x80000000u);
}
static __device__ __forceinline__ float funord(unsigned ou) {
    union { float f; unsigned u; } c;
    c.u = (ou >> 31) ? (ou ^ 0x80000000u) : ~ou;
    return c.f;
}
static __device__ __forceinline__ unsigned long long umax64(unsigned long long a, unsigned long long b) {
    return a > b ? a : b;
}
static __device__ __forceinline__ unsigned long long umin64(unsigned long long a, unsigned long long b) {
    return a < b ? a : b;
}

// ---------------- Kernel 1: LayerNorm expert table -> bf16 [64][2048] ----------------
__global__ void __launch_bounds__(256)
expert_ln_bf16(const float* __restrict__ emb, unsigned short* __restrict__ ebf) {
    const int e = blockIdx.x;          // expert row
    const int t = threadIdx.x;         // 0..255
    const float* row = emb + (size_t)e * D;
    float vals[8];
    float s = 0.f, s2 = 0.f;
#pragma unroll
    for (int i = 0; i < 8; ++i) {
        float v = row[t + 256 * i];
        vals[i] = v; s += v; s2 += v * v;
    }
#pragma unroll
    for (int off = 16; off > 0; off >>= 1) {
        s  += __shfl_xor(s,  off, 32);
        s2 += __shfl_xor(s2, off, 32);
    }
    __shared__ float ws1[8], ws2[8];
    const int wave = t >> 5, lane = t & 31;
    if (lane == 0) { ws1[wave] = s; ws2[wave] = s2; }
    __syncthreads();
    if (t == 0) {
        float a = 0.f, b = 0.f;
        for (int i = 0; i < 8; ++i) { a += ws1[i]; b += ws2[i]; }
        float mu  = a / (float)D;
        float var = b / (float)D - mu * mu;
        ws1[0] = mu; ws2[0] = __frsqrt_rn(var + LN_EPS);
    }
    __syncthreads();
    const float mu = ws1[0], rs = ws2[0];
#pragma unroll
    for (int i = 0; i < 8; ++i)
        ebf[(size_t)e * D + t + 256 * i] = f2bf((vals[i] - mu) * rs);
}

// ---------------- Kernel 2: LN tokens -> bf16 LDS tile -> WMMA sim -> top2+softmax ----
__global__ void __launch_bounds__(128)
router_main(const float* __restrict__ x, const unsigned short* __restrict__ ebf,
            int* __restrict__ idx_out, float* __restrict__ gate_out) {
    __shared__ __align__(16) unsigned short A[16 * APITCH]; // 16 tokens x 2048 bf16, padded
    __shared__ float SIM[16 * NEXP];                        // 16 x 64 f32

    const int tid  = threadIdx.x;
    const int wave = tid >> 5;
    const int lane = tid & 31;
    const int half = lane >> 4;   // 0: lanes 0-15, 1: lanes 16-31
    const int l16  = lane & 15;
    const int tok0 = blockIdx.x * 16;

    // ---- Phase A: LayerNorm 16 tokens (4 per wave), pack bf16 into LDS ----
    for (int i = 0; i < 4; ++i) {
        const int lt = wave * 4 + i;
        const float4* row = (const float4*)(x + (size_t)(tok0 + lt) * D);
        float4 v[16];
        float s = 0.f, s2 = 0.f;
#pragma unroll
        for (int j = 0; j < 16; ++j) {                  // coalesced: 32 lanes * 16B
            float4 q = row[j * 32 + lane];
            v[j] = q;
            s  += q.x + q.y + q.z + q.w;
            s2 += q.x*q.x + q.y*q.y + q.z*q.z + q.w*q.w;
        }
#pragma unroll
        for (int off = 16; off > 0; off >>= 1) {
            s  += __shfl_xor(s,  off, 32);
            s2 += __shfl_xor(s2, off, 32);
        }
        const float mu  = s / (float)D;
        const float var = s2 / (float)D - mu * mu;
        const float rs  = __frsqrt_rn(var + LN_EPS);
#pragma unroll
        for (int j = 0; j < 16; ++j) {
            const float4 q = v[j];
            unsigned p0 = (unsigned)f2bf((q.x - mu) * rs) | ((unsigned)f2bf((q.y - mu) * rs) << 16);
            unsigned p1 = (unsigned)f2bf((q.z - mu) * rs) | ((unsigned)f2bf((q.w - mu) * rs) << 16);
            *(uint2*)&A[lt * APITCH + (j * 32 + lane) * 4] = make_uint2(p0, p1);
        }
    }
    __syncthreads();

    // ---- Phase B: wave w handles experts [16w, 16w+16); 64 WMMAs over K ----
    union Frag { uint4 q[2]; v16bf v; };
    v8f acc = {};
    const uint4* Bbase =
        (const uint4*)(ebf + (size_t)(wave * 16 + l16) * D + 8 * half);
    const unsigned short* Arow = &A[l16 * APITCH + 8 * half];
#pragma unroll 4
    for (int k = 0; k < D; k += 32) {
        Frag a, b;
        // A 16x32 bf16 frag: lanes0-15 K=k+0..7 / k+16..23 ; lanes16-31 +8
        a.q[0] = *(const uint4*)(Arow + k);
        a.q[1] = *(const uint4*)(Arow + k + 16);
        // B 32x16 bf16 frag: lane n = expert col, K=k+0..7 / k+16..23 ; lanes16-31 +8
        b.q[0] = Bbase[k >> 3];
        b.q[1] = Bbase[(k >> 3) + 2];
        acc = __builtin_amdgcn_wmma_f32_16x16x32_bf16(
                  false, a.v, false, b.v, (short)0, acc, false, false);
    }

    // ---- Phase C: spill sim tile (VGPR r -> M = r + 8*half, lane -> N) ----
#pragma unroll
    for (int r = 0; r < 8; ++r)
        SIM[(r + 8 * half) * NEXP + wave * 16 + l16] = acc[r];
    __syncthreads();

    // ---- Phase D: branchless parallel top-2 + softmax(temp = sqrt(D)) ----
    // 8 threads per token, 8 experts each; 64-bit sortable keys:
    //   key = (ord(value) << 32) | (63 - e)   -> distinct keys; ties pick lowest e (jax rule)
    {
        const int t   = tid >> 3;          // token 0..15
        const int sub = tid & 7;           // expert subgroup within token
        const float* srow = &SIM[t * NEXP + sub * 8];
        unsigned long long k1 = 0ull, k2 = 0ull;
#pragma unroll
        for (int j = 0; j < 8; ++j) {
            const int e = sub * 8 + j;
            unsigned long long k =
                ((unsigned long long)ford(srow[j]) << 32) |
                (unsigned long long)(NEXP - 1 - e);
            unsigned long long nk2 = (k > k2) ? k : k2;
            k2 = (k > k1) ? k1 : nk2;
            k1 = (k > k1) ? k  : k1;
        }
        // merge subgroup partial top-2s within the wave (8 lanes per token)
#pragma unroll
        for (int off = 4; off > 0; off >>= 1) {
            unsigned long long o1 = (unsigned long long)
                __shfl_xor((long long)k1, off, 32);
            unsigned long long o2 = (unsigned long long)
                __shfl_xor((long long)k2, off, 32);
            unsigned long long n1 = umax64(k1, o1);
            unsigned long long n2 = umax64(umin64(k1, o1), umax64(k2, o2));
            k1 = n1; k2 = n2;
        }
        if (sub == 0) {
            const int   i1 = NEXP - 1 - (int)(k1 & 63ull);
            const int   i2 = NEXP - 1 - (int)(k2 & 63ull);
            const float m1 = funord((unsigned)(k1 >> 32));
            const float m2 = funord((unsigned)(k2 >> 32));
            const float invT = 0.0220970869120796f;   // 1/sqrt(2048)
            float e2 = __expf((m2 - m1) * invT);      // <= 1
            float g1 = 1.f / (1.f + e2);
            const int tok = tok0 + t;
            idx_out[tok * 2 + 0]  = i1;
            idx_out[tok * 2 + 1]  = i2;
            gate_out[tok * 2 + 0] = g1;
            gate_out[tok * 2 + 1] = e2 * g1;          // = 1 - g1, computed stably
        }
    }
}

extern "C" void kernel_launch(void* const* d_in, const int* in_sizes, int n_in,
                              void* d_out, int out_size, void* d_ws, size_t ws_size,
                              hipStream_t stream) {
    const float* x   = (const float*)d_in[0];   // [4,4096,2048] f32
    const float* emb = (const float*)d_in[1];   // [64,2048] f32
    unsigned short* ebf = (unsigned short*)d_ws;        // 64*2048 bf16 = 256 KB scratch
    int*   idx_out  = (int*)d_out;                      // [16384,2] int32
    float* gate_out = (float*)d_out + (size_t)NTOK * 2; // [16384,2] f32

    expert_ln_bf16<<<NEXP, 256, 0, stream>>>(emb, ebf);
    router_main<<<NTOK / 16, 128, 0, stream>>>(x, ebf, idx_out, gate_out);
}
